// Text_Generation_Net_36507222016364
// MI455X (gfx1250) — compile-verified, matrix operand order checked by reading, not run on hardware
//
#include <hip/hip_runtime.h>
#include <hip/hip_bf16.h>
#include <stdint.h>

// ---------------------------------------------------------------------------
// Text generation net: embed -> LSTM(T=256) -> vocab projection
// B=16, T=256, H=512, V=50257, G=4H=2048
//   * GEMMs in f16 with f32 accumulation via v_wmma_f32_16x16x32_f16.
//   * Register blocking: each wave owns a 16x64 output strip (4 accumulators)
//     so one A fragment feeds 4 WMMAs (~13 FLOP/byte from cache).
//   * LSTM: 16 persistent WGs, cell state in LDS, double-buffered f16 h,
//     one release/acquire grid barrier per step.
// ---------------------------------------------------------------------------

#define BN 16
#define TN 256
#define HN 512
#define VN 50257
#define GN 2048            // 4*H
#define VPAD 50688         // 99*512: multiple of the 512-wide WG tile
#define NWG_LSTM 16

typedef __attribute__((ext_vector_type(16))) _Float16 v16h;
typedef __attribute__((ext_vector_type(8)))  _Float16 h8;
typedef __attribute__((ext_vector_type(8)))  float    v8f;
typedef __attribute__((ext_vector_type(4)))  float    f4;

// ---------------------------------------------------------------------------
// WMMA fragment loaders.
// A-matrix 16x32 f16 layout (CDNA5 ISA 7.12.2):
//   lanes 0-15 : M = lane,    elems 0..7 -> K=k0+0..7,  elems 8..15 -> K=k0+16..23
//   lanes 16-31: M = lane-16, elems 0..7 -> K=k0+8..15, elems 8..15 -> K=k0+24..31
// B operand is packed [N][K] so column n stripes exactly like row m of A.
// ---------------------------------------------------------------------------
__device__ __forceinline__ v16h load_frag_f16(const _Float16* __restrict__ row,
                                              int k0, int lane) {
    const int khalf = (lane < 16) ? 0 : 8;
    h8 x0 = *(const h8*)(row + k0 + khalf);
    h8 x1 = *(const h8*)(row + k0 + 16 + khalf);
    v16h a;
#pragma unroll
    for (int i = 0; i < 8; ++i) { a[i] = x0[i]; a[8 + i] = x1[i]; }
    return a;
}

__device__ __forceinline__ v16h load_frag_f32(const float* __restrict__ row,
                                              int k0, int lane) {
    const int khalf = (lane < 16) ? 0 : 8;
    f4 x0 = *(const f4*)(row + k0 + khalf);
    f4 x1 = *(const f4*)(row + k0 + khalf + 4);
    f4 y0 = *(const f4*)(row + k0 + 16 + khalf);
    f4 y1 = *(const f4*)(row + k0 + 16 + khalf + 4);
    v16h a;
#pragma unroll
    for (int i = 0; i < 4; ++i) {
        a[i]      = (_Float16)x0[i];
        a[4 + i]  = (_Float16)x1[i];
        a[8 + i]  = (_Float16)y0[i];
        a[12 + i] = (_Float16)y1[i];
    }
    return a;
}

// ---------------------------------------------------------------------------
// Weight convert + transpose: src [K][Nsrc] f32 -> dst [Npad][K] f16
// Rows n >= nValid are zero-filled (vocab padding).
// ---------------------------------------------------------------------------
__global__ void cvt_transpose_f16(const float* __restrict__ src,
                                  _Float16* __restrict__ dst,
                                  int K, int Nsrc, int Npad, int nValid) {
    long idx = (long)blockIdx.x * blockDim.x + threadIdx.x;
    long total = (long)Npad * K;
    if (idx >= total) return;
    int n = (int)(idx / K);
    int k = (int)(idx % K);
    float v = (n < nValid) ? src[(long)k * Nsrc + n] : 0.0f;
    dst[idx] = (_Float16)v;
}

__global__ void zero_counters_kernel(int* __restrict__ c, int n) {
    for (int i = threadIdx.x; i < n; i += blockDim.x) c[i] = 0;
}

// ---------------------------------------------------------------------------
// Kernel 1: zx[t*B+b][g] = emb[inputs[b,t]] @ kernel + bias  (f32 out)
// grid = (M/16, G/512), block = 256 (8 waves), wave -> 16x64 strip (4 tiles).
// ---------------------------------------------------------------------------
__global__ void zx_gemm_kernel(const int* __restrict__ inputs,
                               const float* __restrict__ emb,
                               const _Float16* __restrict__ kT,   // [G][H]
                               const float* __restrict__ bias,
                               float* __restrict__ zx) {          // [M][G]
    const int lane = threadIdx.x & 31;
    const int w    = threadIdx.x >> 5;
    const int m0   = blockIdx.x * 16;
    const int n0   = (blockIdx.y * 8 + w) * 64;
    const int r    = m0 + (lane & 15);      // zx row = t*B + b
    const int t    = r >> 4;
    const int b    = r & 15;
    const float* arow = emb + (size_t)inputs[b * TN + t] * HN;

    const _Float16* brow[4];
    v8f acc[4];
#pragma unroll
    for (int j = 0; j < 4; ++j) {
        brow[j] = kT + (size_t)(n0 + j * 16 + (lane & 15)) * HN;
        acc[j] = (v8f){};
    }

#pragma unroll 2
    for (int k0 = 0; k0 < HN; k0 += 32) {
        v16h af = load_frag_f32(arow, k0, lane);   // one A frag feeds 4 WMMAs
#pragma unroll
        for (int j = 0; j < 4; ++j) {
            v16h bf = load_frag_f16(brow[j], k0, lane);
            acc[j] = __builtin_amdgcn_wmma_f32_16x16x32_f16(
                false, af, false, bf, (short)0, acc[j], false, false);
        }
    }
    const int hi = lane >> 4;
#pragma unroll
    for (int j = 0; j < 4; ++j) {
        const int col = n0 + j * 16 + (lane & 15);
        const float bv = bias[col];
#pragma unroll
        for (int rr = 0; rr < 8; ++rr)
            zx[(size_t)(m0 + rr + hi * 8) * GN + col] = acc[j][rr] + bv;
    }
}

// ---------------------------------------------------------------------------
// Persistent LSTM kernel: 16 WGs x 256 threads (8 waves).
// WG g owns hidden units j in [g*32, g*32+32) and gate columns
// {j, j+512, j+1024, j+1536}. Wave w -> gate q=w>>1, col half w&1.
// ---------------------------------------------------------------------------
__device__ __forceinline__ void lstm_grid_sync(int* cnt, int tid) {
    __syncthreads();
    if (tid == 0) {
        __hip_atomic_fetch_add(cnt, 1, __ATOMIC_RELEASE, __HIP_MEMORY_SCOPE_AGENT);
        while (__hip_atomic_load(cnt, __ATOMIC_ACQUIRE, __HIP_MEMORY_SCOPE_AGENT)
               < NWG_LSTM) { }
    }
    __syncthreads();
}

__global__ void lstm_kernel(const float* __restrict__ zx,        // [T*B][G]
                            const _Float16* __restrict__ recT,   // [G][H]
                            _Float16* __restrict__ hbuf,         // [2][B][H]
                            _Float16* __restrict__ hs,           // [B][T][H]
                            float* __restrict__ outTail,         // hT,cT (f32)
                            int* __restrict__ counters) {
    __shared__ float zbuf[4][BN][32];
    __shared__ float cbuf[BN][32];

    const int tid   = threadIdx.x;
    const int lane  = tid & 31;
    const int w     = tid >> 5;
    const int g     = blockIdx.x;
    const int jbase = g * 32;

    // Zero own slice of h (buffer 0) and cell state.
    for (int u = tid; u < BN * 32; u += blockDim.x) {
        int b = u >> 5, jj = u & 31;
        cbuf[b][jj] = 0.0f;
        hbuf[b * HN + jbase + jj] = (_Float16)0.0f;
    }
    __threadfence();
    lstm_grid_sync(&counters[0], tid);

    const int q    = w >> 1;          // gate: 0=i 1=f 2=g 3=o
    const int half = w & 1;
    const int n    = q * HN + jbase + half * 16 + (lane & 15);
    const _Float16* brow = recT + (size_t)n * HN;
    const int hi = lane >> 4;

    for (int t = 0; t < TN; ++t) {
        const _Float16* hsrc = hbuf + (size_t)(t & 1) * (BN * HN);
        const _Float16* arow = hsrc + (lane & 15) * HN;

        v8f acc;
#pragma unroll
        for (int rr = 0; rr < 8; ++rr)
            acc[rr] = zx[(size_t)(t * BN + rr + hi * 8) * GN + n];

#pragma unroll 4
        for (int k0 = 0; k0 < HN; k0 += 32) {
            v16h af = load_frag_f16(arow, k0, lane);
            v16h bf = load_frag_f16(brow, k0, lane);
            acc = __builtin_amdgcn_wmma_f32_16x16x32_f16(
                false, af, false, bf, (short)0, acc, false, false);
        }

        const int cc = half * 16 + (lane & 15);
#pragma unroll
        for (int rr = 0; rr < 8; ++rr)
            zbuf[q][rr + hi * 8][cc] = acc[rr];
        __syncthreads();

        _Float16* hdst = hbuf + (size_t)((t + 1) & 1) * (BN * HN);
        for (int u = tid; u < BN * 32; u += blockDim.x) {
            int b = u >> 5, jj = u & 31;
            float zi = zbuf[0][b][jj];
            float zf = zbuf[1][b][jj];
            float zg = zbuf[2][b][jj];
            float zo = zbuf[3][b][jj];
            float iv = 1.0f / (1.0f + __expf(-zi));
            float fv = 1.0f / (1.0f + __expf(-zf));
            float gv = tanhf(zg);
            float ov = 1.0f / (1.0f + __expf(-zo));
            float c  = fv * cbuf[b][jj] + iv * gv;
            cbuf[b][jj] = c;
            float h  = ov * tanhf(c);
            int j = jbase + jj;
            hdst[b * HN + j] = (_Float16)h;
            hs[((size_t)b * TN + t) * HN + j] = (_Float16)h;
            if (t == TN - 1) {
                outTail[b * HN + j] = h;                 // hT
                outTail[BN * HN + b * HN + j] = c;       // cT
            }
        }
        __threadfence();
        lstm_grid_sync(&counters[t + 1], tid);
    }
}

// ---------------------------------------------------------------------------
// Kernel 3: logits = hs @ W_out + b_out   (211 GFLOP, dominant)
// grid = (M/16, VPAD/512), block = 256 (8 waves), wave -> 16x64 strip.
// ---------------------------------------------------------------------------
__global__ void logits_gemm_kernel(const _Float16* __restrict__ hs,     // [M][H]
                                   const _Float16* __restrict__ WoutT,  // [VPAD][H]
                                   const float* __restrict__ b_out,
                                   float* __restrict__ out) {           // [M][V]
    const int lane = threadIdx.x & 31;
    const int w    = threadIdx.x >> 5;
    const int m0   = blockIdx.x * 16;
    const int n0   = (blockIdx.y * 8 + w) * 64;
    const _Float16* arow = hs + (size_t)(m0 + (lane & 15)) * HN;

    const _Float16* brow[4];
    v8f acc[4];
#pragma unroll
    for (int j = 0; j < 4; ++j) {
        brow[j] = WoutT + (size_t)(n0 + j * 16 + (lane & 15)) * HN; // pad rows = 0
        acc[j] = (v8f){};
    }

#pragma unroll 2
    for (int k0 = 0; k0 < HN; k0 += 32) {
        v16h af = load_frag_f16(arow, k0, lane);   // one A frag feeds 4 WMMAs
#pragma unroll
        for (int j = 0; j < 4; ++j) {
            v16h bf = load_frag_f16(brow[j], k0, lane);
            acc[j] = __builtin_amdgcn_wmma_f32_16x16x32_f16(
                false, af, false, bf, (short)0, acc[j], false, false);
        }
    }
    const int hi = lane >> 4;
#pragma unroll
    for (int j = 0; j < 4; ++j) {
        const int col = n0 + j * 16 + (lane & 15);
        if (col < VN) {
            const float bv = b_out[col];
#pragma unroll
            for (int rr = 0; rr < 8; ++rr)
                out[(size_t)(m0 + rr + hi * 8) * VN + col] = acc[j][rr] + bv;
        }
    }
}

// ---------------------------------------------------------------------------
// Host-side launch
// ---------------------------------------------------------------------------
extern "C" void kernel_launch(void* const* d_in, const int* in_sizes, int n_in,
                              void* d_out, int out_size, void* d_ws, size_t ws_size,
                              hipStream_t stream) {
    (void)in_sizes; (void)n_in; (void)out_size; (void)ws_size;
    const int*   inputs     = (const int*)  d_in[0];   // [B,T]
    const float* emb_table  = (const float*)d_in[1];   // [V,H]
    const float* kernel_w   = (const float*)d_in[2];   // [H,4H]
    const float* rec_kernel = (const float*)d_in[3];   // [H,4H]
    const float* bias       = (const float*)d_in[4];   // [4H]
    const float* W_out      = (const float*)d_in[5];   // [H,V]
    const float* b_out      = (const float*)d_in[6];   // [V]

    float* logits  = (float*)d_out;                              // [B*T,V]
    float* outTail = (float*)d_out + (size_t)BN * TN * VN;       // hT,cT

    // Workspace carve (256B aligned blocks)
    uint8_t* p = (uint8_t*)d_ws;
    auto carve = [&](size_t bytes) {
        void* r = (void*)p;
        p += (bytes + 255) & ~(size_t)255;
        return r;
    };
    _Float16* kT    = (_Float16*)carve((size_t)GN   * HN * 2);   // [G][H]
    _Float16* recT  = (_Float16*)carve((size_t)GN   * HN * 2);   // [G][H]
    _Float16* WoutT = (_Float16*)carve((size_t)VPAD * HN * 2);   // [VPAD][H]
    float*    zx    = (float*)   carve((size_t)TN * BN * GN * 4);// [T*B][G]
    _Float16* hsbuf = (_Float16*)carve((size_t)BN * TN * HN * 2);// [B][T][H]
    _Float16* hbuf  = (_Float16*)carve((size_t)2 * BN * HN * 2); // double-buffered h
    int*      cnts  = (int*)     carve((size_t)(TN + 2) * 4);

    // 0) zero the grid-barrier counters
    zero_counters_kernel<<<1, 256, 0, stream>>>(cnts, TN + 2);

    // 1) weight converts (transpose to [N][K] f16)
    {
        long tot = (long)GN * HN;
        cvt_transpose_f16<<<(tot + 255) / 256, 256, 0, stream>>>(
            kernel_w, kT, HN, GN, GN, GN);
        cvt_transpose_f16<<<(tot + 255) / 256, 256, 0, stream>>>(
            rec_kernel, recT, HN, GN, GN, GN);
        long totv = (long)VPAD * HN;
        cvt_transpose_f16<<<(totv + 255) / 256, 256, 0, stream>>>(
            W_out, WoutT, HN, VN, VPAD, VN);
    }

    // 2) embed + input projection: zx = x @ kernel + bias
    {
        dim3 grid(TN * BN / 16, GN / 512);
        zx_gemm_kernel<<<grid, 256, 0, stream>>>(inputs, emb_table, kT, bias, zx);
    }

    // 3) recurrence (persistent, 16 WGs, 1 grid barrier per step)
    lstm_kernel<<<NWG_LSTM, 256, 0, stream>>>(zx, recT, hbuf, hsbuf, outTail, cnts);

    // 4) vocab projection: logits = hs @ W_out + b_out
    {
        dim3 grid(TN * BN / 16, VPAD / 512);
        logits_gemm_kernel<<<grid, 256, 0, stream>>>(hsbuf, WoutT, b_out, logits);
    }
}
